// SSIM_773094113841
// MI455X (gfx1250) — matrix operand order repaired and checked by reading, not compiled, not run on hardware
//
#include <hip/hip_runtime.h>
#include <hip/hip_bf16.h>

typedef _Float16 half_t;
typedef __attribute__((ext_vector_type(2)))  _Float16 h2;
typedef __attribute__((ext_vector_type(4)))  _Float16 v4h;
typedef __attribute__((ext_vector_type(8)))  _Float16 v8h;
typedef __attribute__((ext_vector_type(16))) _Float16 v16h;
typedef __attribute__((ext_vector_type(8)))  float    v8f;

#define IMG    512
#define OUTW   502          // 512 - 11 + 1
#define TPD    32           // tiles per dimension (32*16 = 512 >= 502)
#define WPB    4            // waves per block
#define RSTRH  36           // f16 halo row stride (72 B: 8B-aligned rows, conflict-free gathers)
#define KSTR   40           // f16 H column stride (80 B: 16B-aligned, conflict-free b128)
#define NBINS  256

union V16 {                 // v16h element e <-> (VGPR e/2, half e%2): matches ISA A/B layouts
    v16h v;
    h2   p[8];
    v8h  q[2];
};

__global__ void __launch_bounds__(256) ssim_init_kernel(float* bins) {
    bins[threadIdx.x] = 0.0f;
}

__global__ void __launch_bounds__(256) ssim_finalize_kernel(const float* __restrict__ bins,
                                                            float* __restrict__ out, float inv) {
    __shared__ float sm[8];
    const int tid = threadIdx.x;
    float v = bins[tid];
#pragma unroll
    for (int off = 16; off > 0; off >>= 1) v += __shfl_xor(v, off, 32);
    if ((tid & 31) == 0) sm[tid >> 5] = v;
    __syncthreads();
    if (tid < 8) {
        float t = sm[tid];
#pragma unroll
        for (int off = 4; off > 0; off >>= 1) t += __shfl_xor(t, off, 8);
        if (tid == 0) out[0] = t * inv;
    }
}

__global__ void __launch_bounds__(128) ssim_wmma_kernel(
    const float* __restrict__ X, const float* __restrict__ Y,
    const float* __restrict__ win, float* __restrict__ bins)
{
    // Per-wave LDS staging (waves independent; no cross-wave barriers needed).
    __shared__ __align__(16) half_t ldsX[WPB][32 * RSTRH];
    __shared__ __align__(16) half_t ldsY[WPB][32 * RSTRH];
    __shared__ __align__(16) half_t ldsH[WPB][5][16 * KSTR];   // column-major: [n][k]
    __shared__ float wpad[47];   // zero-padded weights: wpad[15 + j] = w[j], j in [0,10]

    const int tid  = threadIdx.x;
    const int lane = tid & 31;
    const int wave = tid >> 5;
    const int lhi  = lane >> 4;   // lane half
    const int llo  = lane & 15;

    if (tid < 47) wpad[tid] = 0.0f;
    __syncthreads();
    if (tid < 11) wpad[15 + tid] = win[tid];
    __syncthreads();

    const int img     = blockIdx.y;
    const int tileLin = blockIdx.x * WPB + wave;       // 0..1023
    const int ty      = tileLin >> 5;
    const int tx      = tileLin & 31;
    const int r0      = ty * 16;
    const int c0      = tx * 16;

    const float* Xi = X + (size_t)img * IMG * IMG;
    const float* Yi = Y + (size_t)img * IMG * IMG;

    // ---- Constant Toeplitz operands (contiguous k order, same for A and B) ----
    // Horizontal B (32x16): B[k][n] = w[k-n]; lane: n=llo, element e -> k = lhi*16 + e.
    // Vertical   A (16x32): A[m][k] = w[k-m]; lane: m=llo, element e -> k = lhi*16 + e.
    v16h bW, aW;
#pragma unroll
    for (int e = 0; e < 16; ++e) {
        const int kb = lhi * 16 + e;
        bW[e] = (half_t)wpad[15 + kb - llo];
        const int ka = lhi * 16 + e;
        aW[e] = (half_t)wpad[15 + ka - llo];
    }

    // ---- Stage 32x32 halo of X and Y as f16 (float4 global loads, b64 LDS stores).
    //      Clamp-to-edge keeps fringe finite; fringe outputs masked from the mean. ----
    half_t* rX = ldsX[wave];
    half_t* rY = ldsY[wave];
    {
        const int rsub = lane >> 3;            // 0..3
        const int colv = (lane & 7) * 4;       // 0,4,...,28
        int cs = c0 + colv; cs = cs < (IMG - 4) ? cs : (IMG - 4);
#pragma unroll
        for (int i = 0; i < 8; ++i) {
            const int row = i * 4 + rsub;
            int rr = r0 + row; rr = rr < (IMG - 1) ? rr : (IMG - 1);
            const float4 xv = *reinterpret_cast<const float4*>(Xi + (size_t)rr * IMG + cs);
            const float4 yv = *reinterpret_cast<const float4*>(Yi + (size_t)rr * IMG + cs);
            v4h xh, yh;
            xh[0] = (half_t)xv.x; xh[1] = (half_t)xv.y; xh[2] = (half_t)xv.z; xh[3] = (half_t)xv.w;
            yh[0] = (half_t)yv.x; yh[1] = (half_t)yv.y; yh[2] = (half_t)yv.z; yh[3] = (half_t)yv.w;
            *reinterpret_cast<v4h*>(&rX[row * RSTRH + colv]) = xh;
            *reinterpret_cast<v4h*>(&rY[row * RSTRH + colv]) = yh;
        }
    }

    half_t* H0 = ldsH[wave][0];
    half_t* H1 = ldsH[wave][1];
    half_t* H2 = ldsH[wave][2];
    half_t* H3 = ldsH[wave][3];
    half_t* H4 = ldsH[wave][4];

    // ---- Horizontal pass: 2 row-blocks x 5 streams, one WMMA each ----
#pragma unroll
    for (int b = 0; b < 2; ++b) {
        V16 ax, ay, axx, ayy, axy;
        const int row = b * 16 + llo;
#pragma unroll
        for (int j = 0; j < 8; ++j) {                      // element pair (2j, 2j+1)
            const int e = 2 * j;
            const int k = lhi * 16 + e;
            const h2 xp = *reinterpret_cast<const h2*>(&rX[row * RSTRH + k]);
            const h2 yp = *reinterpret_cast<const h2*>(&rY[row * RSTRH + k]);
            ax.p[j]  = xp;
            ay.p[j]  = yp;
            axx.p[j] = xp * xp;                            // v_pk_mul_f16
            ayy.p[j] = yp * yp;
            axy.p[j] = xp * yp;
        }
        const v8f z = {};
        v8f hx  = __builtin_amdgcn_wmma_f32_16x16x32_f16(false, ax.v,  false, bW, (short)0, z, false, false);
        v8f hy  = __builtin_amdgcn_wmma_f32_16x16x32_f16(false, ay.v,  false, bW, (short)0, z, false, false);
        v8f hxx = __builtin_amdgcn_wmma_f32_16x16x32_f16(false, axx.v, false, bW, (short)0, z, false, false);
        v8f hyy = __builtin_amdgcn_wmma_f32_16x16x32_f16(false, ayy.v, false, bW, (short)0, z, false, false);
        v8f hxy = __builtin_amdgcn_wmma_f32_16x16x32_f16(false, axy.v, false, bW, (short)0, z, false, false);

        // D layout: VGPR v -> row M = v + lhi*8 (8 consecutive rows), col N = llo.
        // Column-major H => one 16B store per stream.
        const int kbase = b * 16 + lhi * 8;
        v8h px, py, pxx, pyy, pxy;
#pragma unroll
        for (int v = 0; v < 8; ++v) {
            px[v]  = (half_t)hx[v];
            py[v]  = (half_t)hy[v];
            pxx[v] = (half_t)hxx[v];
            pyy[v] = (half_t)hyy[v];
            pxy[v] = (half_t)hxy[v];
        }
        *reinterpret_cast<v8h*>(&H0[llo * KSTR + kbase]) = px;
        *reinterpret_cast<v8h*>(&H1[llo * KSTR + kbase]) = py;
        *reinterpret_cast<v8h*>(&H2[llo * KSTR + kbase]) = pxx;
        *reinterpret_cast<v8h*>(&H3[llo * KSTR + kbase]) = pyy;
        *reinterpret_cast<v8h*>(&H4[llo * KSTR + kbase]) = pxy;
    }

    // ---- Vertical pass: B gather = two b128 loads per stream ----
    const int gbase = llo * KSTR + lhi * 16;   // B: row K = lhi*16 + e, col N = llo
    V16 bx, by, bxx, byy, bxy;
    bx.q[0]  = *reinterpret_cast<const v8h*>(&H0[gbase]);
    bx.q[1]  = *reinterpret_cast<const v8h*>(&H0[gbase + 8]);
    by.q[0]  = *reinterpret_cast<const v8h*>(&H1[gbase]);
    by.q[1]  = *reinterpret_cast<const v8h*>(&H1[gbase + 8]);
    bxx.q[0] = *reinterpret_cast<const v8h*>(&H2[gbase]);
    bxx.q[1] = *reinterpret_cast<const v8h*>(&H2[gbase + 8]);
    byy.q[0] = *reinterpret_cast<const v8h*>(&H3[gbase]);
    byy.q[1] = *reinterpret_cast<const v8h*>(&H3[gbase + 8]);
    bxy.q[0] = *reinterpret_cast<const v8h*>(&H4[gbase]);
    bxy.q[1] = *reinterpret_cast<const v8h*>(&H4[gbase + 8]);

    const v8f z = {};
    v8f mu1 = __builtin_amdgcn_wmma_f32_16x16x32_f16(false, aW, false, bx.v,  (short)0, z, false, false);
    v8f mu2 = __builtin_amdgcn_wmma_f32_16x16x32_f16(false, aW, false, by.v,  (short)0, z, false, false);
    v8f exx = __builtin_amdgcn_wmma_f32_16x16x32_f16(false, aW, false, bxx.v, (short)0, z, false, false);
    v8f eyy = __builtin_amdgcn_wmma_f32_16x16x32_f16(false, aW, false, byy.v, (short)0, z, false, false);
    v8f exy = __builtin_amdgcn_wmma_f32_16x16x32_f16(false, aW, false, bxy.v, (short)0, z, false, false);

    // ---- SSIM map: single v_rcp_f32 per element (1-ulp approx; result is a mean
    //      over 8M pixels, so the approximation error is far below fp32 noise).
    //      Select-based masking keeps fringe Inf/NaN out of the sum. ----
    const float C1 = 0.01f * 0.01f;
    const float C2 = 0.03f * 0.03f;
    float s = 0.0f;
    const int ocol = c0 + llo;
#pragma unroll
    for (int v = 0; v < 8; ++v) {
        const int orow = r0 + v + lhi * 8;
        const float m1 = mu1[v], m2 = mu2[v];
        const float m1s = m1 * m1, m2s = m2 * m2, m12 = m1 * m2;
        const float s1  = exx[v] - m1s;
        const float s2  = eyy[v] - m2s;
        const float s12 = exy[v] - m12;
        const float num = (2.0f * m12 + C1) * (2.0f * s12 + C2);
        const float den = (m1s + m2s + C1) * (s1 + s2 + C2);
        const float val = num * __builtin_amdgcn_rcpf(den);
        s += (orow < OUTW && ocol < OUTW) ? val : 0.0f;
    }

    // Wave32 reduction, then one atomic per wave into 256 L2 bins
    // (avoids 32k same-address atomics serializing on one L2 atomic unit).
#pragma unroll
    for (int off = 16; off > 0; off >>= 1)
        s += __shfl_xor(s, off, 32);
    if (lane == 0)
        atomicAdd(&bins[(blockIdx.x + 33 * blockIdx.y + wave * 64) & (NBINS - 1)], s);
}

extern "C" void kernel_launch(void* const* d_in, const int* in_sizes, int n_in,
                              void* d_out, int out_size, void* d_ws, size_t ws_size,
                              hipStream_t stream) {
    const float* X   = (const float*)d_in[0];
    const float* Y   = (const float*)d_in[1];
    const float* win = (const float*)d_in[2];
    float* out  = (float*)d_out;
    float* bins = (float*)d_ws;

    const int N = in_sizes[0] / (IMG * IMG);           // batch size (32)
    const float inv = 1.0f / ((float)N * (float)OUTW * (float)OUTW);

    ssim_init_kernel<<<1, NBINS, 0, stream>>>(bins);

    dim3 grid((TPD * TPD) / WPB, N);                   // 256 x N blocks, 4 waves each
    ssim_wmma_kernel<<<grid, 128, 0, stream>>>(X, Y, win, bins);

    ssim_finalize_kernel<<<1, NBINS, 0, stream>>>(bins, out, inv);
}
